// ContextEncoder_37400575213690
// MI455X (gfx1250) — matrix operand order repaired and checked by reading, not compiled, not run on hardware
//
#include <hip/hip_runtime.h>
#include <hip/hip_bf16.h>

// 2-layer tanh RNN for MI455X (gfx1250, wave32, WMMA).
// B=256, T=1024, I=19, H=128.
// Pipeline (intermediates in [T,B,H] so each scan step reads a contiguous 8KB tile):
//   prep:   f32->f16 weight conversion, bias folding
//   proj0:  pre0[t,b,:] = x[b,t,:] @ W_ih0^T + (b_ih0+b_hh0)     (K=19, VALU)
//   scan0:  h_t = tanh(pre0_t + h @ W_hh0^T) -> out1 [T,B,H] f16 (WMMA, LDS ping-pong)
//   gemm1:  pre1 = out1 @ W_ih1^T + (b_ih1+b_hh1)                (WMMA, full parallel)
//   scan1:  h_t = tanh(pre1_t + h @ W_hh1^T) -> d_out [B,T,H] f32

#define RNN_B 256
#define RNN_T 1024
#define RNN_H 128
#define RNN_I 19
#define RNN_BT (RNN_B * RNN_T)

typedef __attribute__((ext_vector_type(16))) _Float16 v16h;
typedef __attribute__((ext_vector_type(8)))  _Float16 v8h;
typedef __attribute__((ext_vector_type(8)))  float    v8f;

__device__ __forceinline__ float fast_tanh(float x) {
#if __has_builtin(__builtin_amdgcn_tanhf)
  return __builtin_amdgcn_tanhf(x);           // gfx1250 v_tanh_f32 (TRANS unit)
#elif __has_builtin(__builtin_amdgcn_tanh_f32)
  return __builtin_amdgcn_tanh_f32(x);
#else
  return tanhf(x);
#endif
}

// ---------- prep: convert weights to f16 [n][k], fold biases ----------
__global__ void rnn_prep(const float* __restrict__ Whh0, const float* __restrict__ Whh1,
                         const float* __restrict__ Wih1,
                         const float* __restrict__ bih0, const float* __restrict__ bhh0,
                         const float* __restrict__ bih1, const float* __restrict__ bhh1,
                         _Float16* __restrict__ Whh0h, _Float16* __restrict__ Whh1h,
                         _Float16* __restrict__ Wih1h,
                         float* __restrict__ bias0, float* __restrict__ bias1) {
  int i = blockIdx.x * blockDim.x + threadIdx.x;
  if (i < RNN_H * RNN_H) {
    Whh0h[i] = (_Float16)Whh0[i];
    Whh1h[i] = (_Float16)Whh1[i];
    Wih1h[i] = (_Float16)Wih1[i];
  }
  if (i < RNN_H) {
    bias0[i] = bih0[i] + bhh0[i];
    bias1[i] = bih1[i] + bhh1[i];
  }
}

// ---------- layer0 input projection: K=19 skinny GEMM, VALU ----------
// x is [B,T,I]; pre0 written [T,B,H].
__global__ void rnn_proj0(const float* __restrict__ x,
                          const float* __restrict__ Wih0,  // [128, 19]
                          const float* __restrict__ bias0, // [128]
                          float* __restrict__ pre0) {      // [T,B,H]
  size_t gid = (size_t)blockIdx.x * blockDim.x + threadIdx.x;
  int    h   = (int)(gid & (RNN_H - 1));
  size_t tb  = gid >> 7;
  int    b   = (int)(tb & (RNN_B - 1));
  int    t   = (int)(tb >> 8);
  const float* xr = x + ((size_t)b * RNN_T + t) * RNN_I;
  const float* wr = Wih0 + h * RNN_I;
  float acc = bias0[h];
#pragma unroll
  for (int i = 0; i < RNN_I; ++i) acc = fmaf(xr[i], wr[i], acc);
  pre0[gid] = acc;
}

// ---------- fragment helpers (layouts per ISA 7.12.2, wave32) ----------
// B-matrix (32x16 f16) from W stored [n][k]: lane = col n, halves 0-7 = K kbase..kbase+7,
// halves 8-15 = K kbase+16..kbase+23 (kbase = 32*frag + 8*(lane>=16)).
__device__ __forceinline__ v16h load_bfrag(const _Float16* __restrict__ Wnk,
                                           int n, int kbase) {
  v8h lo = *(const v8h*)(Wnk + (size_t)n * RNN_H + kbase);
  v8h hi = *(const v8h*)(Wnk + (size_t)n * RNN_H + kbase + 16);
  v16h b;
#pragma unroll
  for (int j = 0; j < 8; ++j) { b[j] = lo[j]; b[8 + j] = hi[j]; }
  return b;
}

// A-matrix (16x32 f16) from row-major [m][128]: halves 0-7 = K k0..k0+7,
// halves 8-15 = K k0+16..k0+23 (k0 = 32*frag + 8*(lane>=16)).
__device__ __forceinline__ v16h load_afrag(const _Float16* __restrict__ row, int k0) {
  v8h lo = *(const v8h*)(row + k0);
  v8h hi = *(const v8h*)(row + k0 + 16);
  v16h a;
#pragma unroll
  for (int j = 0; j < 8; ++j) { a[j] = lo[j]; a[8 + j] = hi[j]; }
  return a;
}

// ---------- layer1 input projection: out1[BT,128] @ Wih1^T, WMMA ----------
// Rows indexed by (t*B + b); both out1 and pre1 are [T,B,H], so plain row GEMM.
__global__ __launch_bounds__(256) void
rnn_gemm1(const _Float16* __restrict__ A,    // out1 [BT,128] f16
          const _Float16* __restrict__ Wnk,  // Wih1 f16 [n][k]
          const float* __restrict__ bias,    // [128]
          float* __restrict__ pre1) {        // [BT,128]
  const int wave  = threadIdx.x >> 5;
  const int lane  = threadIdx.x & 31;
  const int lhalf = lane & 15;
  const int hi16  = lane >> 4;
  const int m0    = blockIdx.x * 16;
  const int n     = wave * 16 + lhalf;

  v16h bfrag[4], afrag[4];
  const _Float16* arow = A + (size_t)(m0 + lhalf) * RNN_H;
#pragma unroll
  for (int f = 0; f < 4; ++f) {
    bfrag[f] = load_bfrag(Wnk, n, f * 32 + hi16 * 8);
    afrag[f] = load_afrag(arow, f * 32 + hi16 * 8);
  }

  float bv = bias[n];
  v8f c0, c1;
#pragma unroll
  for (int r = 0; r < 8; ++r) { c0[r] = bv; c1[r] = 0.0f; }

  // Two independent accumulation chains to hide WMMA->WMMA RAW hazard NOPs.
  c0 = __builtin_amdgcn_wmma_f32_16x16x32_f16(false, afrag[0], false, bfrag[0], (short)0, c0, false, false);
  c1 = __builtin_amdgcn_wmma_f32_16x16x32_f16(false, afrag[1], false, bfrag[1], (short)0, c1, false, false);
  c0 = __builtin_amdgcn_wmma_f32_16x16x32_f16(false, afrag[2], false, bfrag[2], (short)0, c0, false, false);
  c1 = __builtin_amdgcn_wmma_f32_16x16x32_f16(false, afrag[3], false, bfrag[3], (short)0, c1, false, false);

#pragma unroll
  for (int r = 0; r < 8; ++r)
    pre1[(size_t)(m0 + r + hi16 * 8) * RNN_H + n] = c0[r] + c1[r];
}

// ---------- recurrent scan: h_t = tanh(pre_t + h @ Whh^T) ----------
// grid = B/16, 256 threads (8 waves, one 16-wide N tile per wave).
// pre is [T,B,H]; C loads for step t+1 are register-pipelined; prefetch 8 steps out.
template <bool WRITE_F32>
__global__ __launch_bounds__(256) void
rnn_scan(const float* __restrict__ pre,     // [T,B,H] f32 (bias folded)
         const _Float16* __restrict__ Whh,  // f16 [n][k]
         _Float16* __restrict__ out16,      // layer0: out1 [T,B,H] f16
         float* __restrict__ out32) {       // layer1: d_out [B,T,H] f32
  __shared__ _Float16 hbuf[2][16 * RNN_H];  // ping-pong hidden state, 8 KB

  const int wave  = threadIdx.x >> 5;
  const int lane  = threadIdx.x & 31;
  const int lhalf = lane & 15;
  const int hi16  = lane >> 4;
  const int b0    = blockIdx.x * 16;
  const int n     = wave * 16 + lhalf;
  const int step  = RNN_B * RNN_H;          // floats per timestep

  // W_hh^T B-fragments resident in VGPRs for the whole scan.
  v16h bfrag[4];
#pragma unroll
  for (int f = 0; f < 4; ++f)
    bfrag[f] = load_bfrag(Whh, n, f * 32 + hi16 * 8);

  // h0 = 0
  for (int i = threadIdx.x; i < 16 * RNN_H; i += 256) hbuf[0][i] = (_Float16)0.0f;
  __syncthreads();

  // C base: row (b0 + hi16*8), col n; row r adds r*H, step adds B*H.
  const float* cptr = pre + (size_t)(b0 + hi16 * 8) * RNN_H + n;

  v8f cnext;
#pragma unroll
  for (int r = 0; r < 8; ++r) cnext[r] = cptr[r * RNN_H];
  cptr += step;

  int cur = 0;
  for (int t = 0; t < RNN_T; ++t) {
    v8f c = cnext;                               // pre(t), loaded last iteration
    if (t + 1 < RNN_T) {                         // issue pre(t+1) loads early
#pragma unroll
      for (int r = 0; r < 8; ++r) cnext[r] = cptr[r * RNN_H];
      if (t + 9 < RNN_T)                         // stream ~1MB ahead into caches
        __builtin_prefetch(cptr + (size_t)8 * step, 0, 1);
      cptr += step;
    }

    // A fragments from LDS hidden state
    v16h a0, a1, a2, a3;
    const _Float16* arow = &hbuf[cur][lhalf * RNN_H];
    a0 = load_afrag(arow, 0 * 32 + hi16 * 8);
    a1 = load_afrag(arow, 1 * 32 + hi16 * 8);
    a2 = load_afrag(arow, 2 * 32 + hi16 * 8);
    a3 = load_afrag(arow, 3 * 32 + hi16 * 8);

    v8f acc;
#pragma unroll
    for (int r = 0; r < 8; ++r) acc[r] = 0.0f;

    // Two interleaved chains -> WMMA hazard slots filled with real work.
    c   = __builtin_amdgcn_wmma_f32_16x16x32_f16(false, a0, false, bfrag[0], (short)0, c,   false, false);
    acc = __builtin_amdgcn_wmma_f32_16x16x32_f16(false, a1, false, bfrag[1], (short)0, acc, false, false);
    c   = __builtin_amdgcn_wmma_f32_16x16x32_f16(false, a2, false, bfrag[2], (short)0, c,   false, false);
    acc = __builtin_amdgcn_wmma_f32_16x16x32_f16(false, a3, false, bfrag[3], (short)0, acc, false, false);

    const int nxt = cur ^ 1;
#pragma unroll
    for (int r = 0; r < 8; ++r) {
      int m = r + hi16 * 8;
      float hv = fast_tanh(c[r] + acc[r]);
      hbuf[nxt][m * RNN_H + n] = (_Float16)hv;
      if (WRITE_F32)   // final layer: required [B,T,H] layout
        out32[((size_t)(b0 + m) * RNN_T + t) * RNN_H + n] = hv;
      else             // layer 0: [T,B,H] for the downstream GEMM
        out16[(size_t)t * step + (size_t)(b0 + m) * RNN_H + n] = (_Float16)hv;
    }
    __syncthreads();
    cur = nxt;
  }
}

extern "C" void kernel_launch(void* const* d_in, const int* in_sizes, int n_in,
                              void* d_out, int out_size, void* d_ws, size_t ws_size,
                              hipStream_t stream) {
  const float* x     = (const float*)d_in[0];
  const float* Wih0  = (const float*)d_in[1];
  const float* Whh0  = (const float*)d_in[2];
  const float* bih0  = (const float*)d_in[3];
  const float* bhh0  = (const float*)d_in[4];
  const float* Wih1  = (const float*)d_in[5];
  const float* Whh1  = (const float*)d_in[6];
  const float* bih1  = (const float*)d_in[7];
  const float* bhh1  = (const float*)d_in[8];
  float* out = (float*)d_out;

  // workspace layout (~336 MB total)
  char* ws = (char*)d_ws;
  _Float16* Whh0h = (_Float16*)(ws + 0);
  _Float16* Whh1h = (_Float16*)(ws + 32768);
  _Float16* Wih1h = (_Float16*)(ws + 65536);
  float*    bias0 = (float*)(ws + 98304);
  float*    bias1 = (float*)(ws + 98816);
  float*    pre0  = (float*)(ws + 102400);                                 // 134 MB f32
  _Float16* out1  = (_Float16*)(ws + 102400 + (size_t)RNN_BT * RNN_H * 4); // 67 MB f16
  float*    pre1  = (float*)(ws + 102400 + (size_t)RNN_BT * RNN_H * 6);    // 134 MB f32

  rnn_prep<<<64, 256, 0, stream>>>(Whh0, Whh1, Wih1, bih0, bhh0, bih1, bhh1,
                                   Whh0h, Whh1h, Wih1h, bias0, bias1);

  rnn_proj0<<<((size_t)RNN_BT * RNN_H) / 256, 256, 0, stream>>>(x, Wih0, bias0, pre0);

  rnn_scan<false><<<RNN_B / 16, 256, 0, stream>>>(pre0, Whh0h, out1, nullptr);

  rnn_gemm1<<<RNN_BT / 16, 256, 0, stream>>>(out1, Wih1h, bias1, pre1);

  rnn_scan<true><<<RNN_B / 16, 256, 0, stream>>>(pre1, Whh1h, nullptr, out);
}
